// MSpatialAB_14096082665903
// MI455X (gfx1250) — matrix operand order repaired and checked by reading, not compiled, not run on hardware
//
#include <hip/hip_runtime.h>
#include <hip/hip_bf16.h>
#include <math.h>

typedef __attribute__((ext_vector_type(16))) _Float16 v16h;
typedef __attribute__((ext_vector_type(8)))  float    v8f;
typedef __attribute__((ext_vector_type(4)))  int      v4i;

#define T_TOKENS 131072   // B*H*W = 2*256*256
#define HH 256
#define WW 256
#define CC 128

// ---------------- ws layout (bytes) ----------------
#define OFF_Y     ((size_t)0)                    // f32 NHWC activation, 64 MiB
#define OFF_LN    ((size_t)67108864)             // f16 [T,128], 32 MiB
#define OFF_QKV   ((size_t)100663296)            // f16 [T,384], 96 MiB
#define OFF_ATT   ((size_t)201326592)            // f16 [T,128], 32 MiB
#define OFF_FFH1  OFF_QKV                        // f16 [T,512], 128 MiB (reuses QKV+ATT)
#define OFF_FFH2  ((size_t)234881024)            // f16 [T,512], 128 MiB
#define OFF_WQKV0 ((size_t)369098752)            // f16 384*128
#define OFF_WQKV1 (OFF_WQKV0 + 98304)
#define OFF_WPRJ0 (OFF_WQKV1 + 98304)            // f16 128*128
#define OFF_WPRJ1 (OFF_WPRJ0 + 32768)
#define OFF_WF1   (OFF_WPRJ1 + 32768)            // f16 512*128
#define OFF_WF2   (OFF_WF1 + 131072)             // f16 128*512

// ---------------- CDNA5 async global->LDS ----------------
#if defined(__has_builtin)
#if __has_builtin(__builtin_amdgcn_global_load_async_to_lds_b128)
#define HAVE_ASYNC_LDS 1
#endif
#if __has_builtin(__builtin_amdgcn_s_wait_asynccnt)
#define HAVE_WAIT_ASYNC 1
#endif
#endif

typedef __attribute__((address_space(1))) v4i* gv4i_p;   // global int4*
typedef __attribute__((address_space(3))) v4i* lv4i_p;   // LDS int4*

__device__ __forceinline__ void async_copy16(const void* g, void* lds) {
#ifdef HAVE_ASYNC_LDS
  __builtin_amdgcn_global_load_async_to_lds_b128(
      (gv4i_p)(uintptr_t)g,
      (lv4i_p)(uint32_t)(uintptr_t)lds, 0, 0);
#else
  *(uint4*)lds = *(const uint4*)g;
#endif
}

__device__ __forceinline__ void async_wait0() {
#ifdef HAVE_ASYNC_LDS
#ifdef HAVE_WAIT_ASYNC
  __builtin_amdgcn_s_wait_asynccnt(0);
#else
  asm volatile("s_wait_asynccnt 0" ::: "memory");
#endif
#endif
}

// ---------------- fragment loaders ----------------
// A fragment (16x32 f16): caller passes base + row*ld + (lane>>4)*8 + k0
__device__ __forceinline__ v16h load_fragA(const _Float16* p) {
  union { uint4 u[2]; v16h h; } f;
  f.u[0] = *reinterpret_cast<const uint4*>(p);
  f.u[1] = *reinterpret_cast<const uint4*>(p + 16);
  return f.h;
}
// B fragment (32x16 f16, stored as Bt[n,k] rows): base + n*ld + (lane>>4)*8 + k0
__device__ __forceinline__ v16h load_fragB(const _Float16* p) {
  union { uint4 u[2]; v16h h; } f;
  f.u[0] = *reinterpret_cast<const uint4*>(p);
  f.u[1] = *reinterpret_cast<const uint4*>(p + 16);
  return f.h;
}

__device__ __forceinline__ float gelu_exact(float v) {
  return 0.5f * v * (1.0f + erff(v * 0.70710678118654752f));
}

// ---------------- layout transforms ----------------
__global__ void k_nchw2nhwc(const float* __restrict__ X, float* __restrict__ Y) {
  size_t i = (size_t)blockIdx.x * blockDim.x + threadIdx.x;   // over B*H*W*C
  int c = (int)(i & 127);
  size_t t = i >> 7;
  int b = (int)(t >> 16);
  int yx = (int)(t & 65535);
  Y[i] = X[(((size_t)(b * CC + c)) << 16) + yx];
}

__global__ void k_nhwc2nchw(const float* __restrict__ Y, float* __restrict__ O) {
  size_t i = (size_t)blockIdx.x * blockDim.x + threadIdx.x;
  int c = (int)(i & 127);
  size_t t = i >> 7;
  int b = (int)(t >> 16);
  int yx = (int)(t & 65535);
  O[(((size_t)(b * CC + c)) << 16) + yx] = Y[i];
}

__global__ void k_cvt16(const float* __restrict__ a, _Float16* __restrict__ o, int n) {
  int i = blockIdx.x * blockDim.x + threadIdx.x;
  if (i < n) o[i] = (_Float16)a[i];
}

// ---------------- LayerNorm (+ optional roll + window partition) ----------------
__global__ __launch_bounds__(256)
void k_ln(const float* __restrict__ Y, const float* __restrict__ g,
          const float* __restrict__ bb, _Float16* __restrict__ out,
          int windowed, int shift) {
  int wv = threadIdx.x >> 5, lane = threadIdx.x & 31;
  int t = blockIdx.x * 8 + wv;   // destination token index
  int src;
  if (windowed) {
    int win = t >> 6, iw = t & 63;
    int b = win >> 10, wy = (win >> 5) & 31, wx = win & 31;
    int ay = (wy << 3) + (iw >> 3), ax = (wx << 3) + (iw & 7);
    int sy = (ay + shift) & 255, sx = (ax + shift) & 255;
    src = ((b << 8) + sy) * 256 + sx;
  } else {
    src = t;
  }
  const float4 v = *((const float4*)(Y + (size_t)src * CC) + lane);
  float s  = v.x + v.y + v.z + v.w;
  float ss = v.x * v.x + v.y * v.y + v.z * v.z + v.w * v.w;
  #pragma unroll
  for (int o = 16; o; o >>= 1) { s += __shfl_xor(s, o, 32); ss += __shfl_xor(ss, o, 32); }
  float mean = s * (1.0f / 128.0f);
  float var  = ss * (1.0f / 128.0f) - mean * mean;
  float inv  = rsqrtf(var + 1e-5f);
  int c = lane << 2;
  _Float16 o4[4];
  o4[0] = (_Float16)((v.x - mean) * inv * g[c + 0] + bb[c + 0]);
  o4[1] = (_Float16)((v.y - mean) * inv * g[c + 1] + bb[c + 1]);
  o4[2] = (_Float16)((v.z - mean) * inv * g[c + 2] + bb[c + 2]);
  o4[3] = (_Float16)((v.w - mean) * inv * g[c + 3] + bb[c + 3]);
  *(uint2*)(out + (size_t)t * CC + c) = *(uint2*)o4;
}

// ---------------- generic WMMA GEMM ----------------
// D[M,N] = A[M,K] @ Bt[N,K]^T (+bias, epilogue). One wave = 16x64 output strip.
// EPI 0: f16 store, scale first 128 cols by 32^-0.5 (QKV)
// EPI 1: f16 store with exact GELU (FF1)
// EPI 2: f32 residual-add into Y with window->NHWC unroll mapping (proj)
// EPI 3: f32 residual-add into Y, raster order (FF2)
template <int EPI>
__global__ __launch_bounds__(256)
void k_gemm(const _Float16* __restrict__ A, const _Float16* __restrict__ Bt,
            const float* __restrict__ bias, _Float16* __restrict__ out16,
            float* __restrict__ Yres, int M, int N, int K, int shift) {
  int wave = (blockIdx.x * blockDim.x + threadIdx.x) >> 5;
  int lane = threadIdx.x & 31;
  int bigN = N >> 6;                      // 64-wide strips
  int tm  = (wave / bigN) << 4;
  int tn0 = (wave % bigN) << 6;
  if (tm >= M) return;
  v8f acc[4] = {};
  const _Float16* arow = A  + (size_t)(tm  + (lane & 15)) * K + ((lane >> 4) << 3);
  const _Float16* brow = Bt + (size_t)(tn0 + (lane & 15)) * K + ((lane >> 4) << 3);
  for (int k0 = 0; k0 < K; k0 += 32) {
    v16h a = load_fragA(arow + k0);
    #pragma unroll
    for (int nt = 0; nt < 4; ++nt) {
      v16h b = load_fragB(brow + (size_t)(nt * 16) * K + k0);
      acc[nt] = __builtin_amdgcn_wmma_f32_16x16x32_f16(false, a, false, b, (short)0,
                                                       acc[nt], false, false);
    }
  }
  int mbase = tm + ((lane >> 4) << 3);
  #pragma unroll
  for (int r = 0; r < 8; ++r) {
    int m = mbase + r;
    size_t rowoff = 0;
    if (EPI == 2) {
      int win = m >> 6, iw = m & 63;
      int b = win >> 10, wy = (win >> 5) & 31, wx = win & 31;
      int ay = (wy << 3) + (iw >> 3), ax = (wx << 3) + (iw & 7);
      int oy = (ay + shift) & 255, ox = (ax + shift) & 255;
      rowoff = (size_t)(((b << 8) + oy) * 256 + ox) * CC;
    }
    #pragma unroll
    for (int nt = 0; nt < 4; ++nt) {
      int n = tn0 + nt * 16 + (lane & 15);
      float v = acc[nt][r] + (bias ? bias[n] : 0.0f);
      if (EPI == 0) {
        if (n < 128) v *= 0.17677669529663687f;   // d^-0.5, d=32
        out16[(size_t)m * N + n] = (_Float16)v;
      } else if (EPI == 1) {
        out16[(size_t)m * N + n] = (_Float16)gelu_exact(v);
      } else if (EPI == 2) {
        Yres[rowoff + n] += v;
      } else {
        Yres[(size_t)m * CC + n] += v;
      }
    }
  }
}

// ---------------- per-window attention ----------------
__device__ __forceinline__ int regseg(int p) {
  return p < (HH - 8) ? 0 : (p < (HH - 4) ? 1 : 2);
}

__global__ __launch_bounds__(256)
void k_attn(const _Float16* __restrict__ QKV, _Float16* __restrict__ OUT,
            const float* __restrict__ btab, int shifted) {
  __shared__ _Float16 sQ[64 * 32];
  __shared__ _Float16 sK[64 * 32];
  __shared__ _Float16 sV[64 * 32];
  __shared__ _Float16 sVT[32 * 64];   // V transposed: [d][token]
  __shared__ float    sS[64 * 64];
  __shared__ _Float16 sP[64 * 64];
  int win = blockIdx.x;
  int tid = threadIdx.x;
  int lane = tid & 31;
  int wv = tid >> 5;
  int wy = (win >> 5) & 31, wx = win & 31;
  int tok = tid >> 2;
  int c = (tid & 3) << 3;

  for (int head = 0; head < 4; ++head) {
    {   // async-stage q,k,v head slices into LDS (each 64x32 f16)
      const _Float16* src = QKV + (size_t)(win * 64 + tok) * 384 + head * 32 + c;
      async_copy16(src,       &sQ[tok * 32 + c]);
      async_copy16(src + 128, &sK[tok * 32 + c]);
      async_copy16(src + 256, &sV[tok * 32 + c]);
      async_wait0();
    }
    __syncthreads();

    // build transposed V while S tiles run
    {
      union { uint4 u; _Float16 h[8]; } vv;
      vv.u = *(uint4*)&sV[tok * 32 + c];
      #pragma unroll
      for (int j = 0; j < 8; ++j) sVT[(c + j) * 64 + tok] = vv.h[j];
    }

    // S = q @ k^T  (64x64, K=32): 16 tiles, 2 per wave
    #pragma unroll
    for (int ii = 0; ii < 2; ++ii) {
      int t = wv * 2 + ii;
      int ti = t >> 2, tj = t & 3;
      v16h a = load_fragA(sQ + (ti * 16 + (lane & 15)) * 32 + ((lane >> 4) << 3));
      v16h b = load_fragB(sK + (tj * 16 + (lane & 15)) * 32 + ((lane >> 4) << 3));
      v8f acc = {};
      acc = __builtin_amdgcn_wmma_f32_16x16x32_f16(false, a, false, b, (short)0, acc,
                                                   false, false);
      int km = tj * 16 + (lane & 15);
      int qb = ti * 16 + ((lane >> 4) << 3);
      int akx = (wx << 3) + (km & 7), aky = (wy << 3) + (km >> 3);
      int lk = shifted ? (regseg(aky) * 3 + regseg(akx)) : 0;
      #pragma unroll
      for (int r = 0; r < 8; ++r) {
        int qn = qb + r;
        int dy = (qn >> 3) - (km >> 3) + 7;
        int dx = (qn & 7) - (km & 7) + 7;
        float v = acc[r] + btab[(dy * 15 + dx) * 4 + head];
        if (shifted) {
          int aqy = (wy << 3) + (qn >> 3), aqx = (wx << 3) + (qn & 7);
          int lq = regseg(aqy) * 3 + regseg(aqx);
          if (lq != lk) v -= 100.0f;
        }
        sS[qn * 64 + km] = v;
      }
    }
    __syncthreads();

    // softmax over rows (64 rows, threads 0..63)
    if (tid < 64) {
      float mx = -1e30f;
      for (int m = 0; m < 64; ++m) mx = fmaxf(mx, sS[tid * 64 + m]);
      float s = 0.0f;
      for (int m = 0; m < 64; ++m) {
        float e = expf(sS[tid * 64 + m] - mx);
        sS[tid * 64 + m] = e;
        s += e;
      }
      float inv = 1.0f / s;
      for (int m = 0; m < 64; ++m) sP[tid * 64 + m] = (_Float16)(sS[tid * 64 + m] * inv);
    }
    __syncthreads();

    // O = P @ V  (64x32, K=64): 8 tiles, 1 per wave, 2 WMMA each.
    // B fragments come from sVT -> contiguous ds_load_b128.
    {
      int ti = wv >> 1, tj = wv & 1;
      int nd = tj * 16 + (lane & 15);
      v8f acc = {};
      #pragma unroll
      for (int k0 = 0; k0 < 64; k0 += 32) {
        v16h a = load_fragA(sP + (ti * 16 + (lane & 15)) * 64 + k0 + ((lane >> 4) << 3));
        v16h b = load_fragB(sVT + nd * 64 + k0 + ((lane >> 4) << 3));
        acc = __builtin_amdgcn_wmma_f32_16x16x32_f16(false, a, false, b, (short)0, acc,
                                                     false, false);
      }
      int mb = ti * 16 + ((lane >> 4) << 3);
      #pragma unroll
      for (int r = 0; r < 8; ++r)
        OUT[(size_t)(win * 64 + mb + r) * CC + head * 32 + nd] = (_Float16)acc[r];
    }
    __syncthreads();
  }
}

// ---------------- depthwise 3x3 + GELU ----------------
__global__ __launch_bounds__(256)
void k_dwconv(const _Float16* __restrict__ Hin, const float* __restrict__ Wd,
              _Float16* __restrict__ Hout) {
  size_t gid = (size_t)blockIdx.x * blockDim.x + threadIdx.x;  // T * 64 threads
  size_t tok = gid >> 6;
  int c = (int)(gid & 63) << 3;
  int b = (int)(tok >> 16);
  int y = (int)(tok >> 8) & 255;
  int x = (int)tok & 255;
  float acc[8] = {0, 0, 0, 0, 0, 0, 0, 0};
  #pragma unroll
  for (int ky = -1; ky <= 1; ++ky) {
    #pragma unroll
    for (int kx = -1; kx <= 1; ++kx) {
      int yy = y + ky, xx = x + kx;
      if (yy < 0 || yy > 255 || xx < 0 || xx > 255) continue;
      const _Float16* p = Hin + (((size_t)(b << 16) + (yy << 8) + xx) << 9) + c;
      union { uint4 u; _Float16 h[8]; } dv;
      dv.u = *(const uint4*)p;
      const float* wp = Wd + ((ky + 1) * 3 + (kx + 1)) * 512 + c;
      #pragma unroll
      for (int j = 0; j < 8; ++j) acc[j] += (float)dv.h[j] * wp[j];
    }
  }
  union { uint4 u; _Float16 h[8]; } ov;
  #pragma unroll
  for (int j = 0; j < 8; ++j) ov.h[j] = (_Float16)gelu_exact(acc[j]);
  *(uint4*)(Hout + (tok << 9) + c) = ov.u;
}

// ---------------- host ----------------
extern "C" void kernel_launch(void* const* d_in, const int* in_sizes, int n_in,
                              void* d_out, int out_size, void* d_ws, size_t ws_size,
                              hipStream_t stream) {
  const float* x        = (const float*)d_in[0];
  const float* w_ln_g   = (const float*)d_in[1];
  const float* w_ln_b   = (const float*)d_in[2];
  const float* w_qkv_w  = (const float*)d_in[3];
  const float* w_qkv_b  = (const float*)d_in[4];
  const float* w_proj_w = (const float*)d_in[5];
  const float* w_proj_b = (const float*)d_in[6];
  const float* w_bias   = (const float*)d_in[7];
  const float* s_ln_g   = (const float*)d_in[8];
  const float* s_ln_b   = (const float*)d_in[9];
  const float* s_qkv_w  = (const float*)d_in[10];
  const float* s_qkv_b  = (const float*)d_in[11];
  const float* s_proj_w = (const float*)d_in[12];
  const float* s_proj_b = (const float*)d_in[13];
  const float* s_bias   = (const float*)d_in[14];
  const float* f_ln_g   = (const float*)d_in[15];
  const float* f_ln_b   = (const float*)d_in[16];
  const float* ff_w1    = (const float*)d_in[17];
  const float* ff_wd    = (const float*)d_in[18];
  const float* ff_w2    = (const float*)d_in[19];

  char* ws = (char*)d_ws;
  float*    Y    = (float*)(ws + OFF_Y);
  _Float16* LN   = (_Float16*)(ws + OFF_LN);
  _Float16* QKV  = (_Float16*)(ws + OFF_QKV);
  _Float16* ATT  = (_Float16*)(ws + OFF_ATT);
  _Float16* FFH1 = (_Float16*)(ws + OFF_FFH1);
  _Float16* FFH2 = (_Float16*)(ws + OFF_FFH2);
  _Float16* WQ0  = (_Float16*)(ws + OFF_WQKV0);
  _Float16* WQ1  = (_Float16*)(ws + OFF_WQKV1);
  _Float16* WP0  = (_Float16*)(ws + OFF_WPRJ0);
  _Float16* WP1  = (_Float16*)(ws + OFF_WPRJ1);
  _Float16* WF1  = (_Float16*)(ws + OFF_WF1);
  _Float16* WF2  = (_Float16*)(ws + OFF_WF2);

  // weight conversion to f16
  k_cvt16<<<(384 * 128 + 255) / 256, 256, 0, stream>>>(w_qkv_w, WQ0, 384 * 128);
  k_cvt16<<<(384 * 128 + 255) / 256, 256, 0, stream>>>(s_qkv_w, WQ1, 384 * 128);
  k_cvt16<<<(128 * 128 + 255) / 256, 256, 0, stream>>>(w_proj_w, WP0, 128 * 128);
  k_cvt16<<<(128 * 128 + 255) / 256, 256, 0, stream>>>(s_proj_w, WP1, 128 * 128);
  k_cvt16<<<(512 * 128 + 255) / 256, 256, 0, stream>>>(ff_w1, WF1, 512 * 128);
  k_cvt16<<<(128 * 512 + 255) / 256, 256, 0, stream>>>(ff_w2, WF2, 128 * 512);

  // NCHW -> NHWC
  k_nchw2nhwc<<<(T_TOKENS * CC) / 256, 256, 0, stream>>>(x, Y);

  const int gLN  = T_TOKENS / 8;
  const int gQKV = (T_TOKENS / 16) * (384 / 64) / 8;  // 6144
  const int gPRJ = (T_TOKENS / 16) * (128 / 64) / 8;  // 2048
  const int gFF1 = (T_TOKENS / 16) * (512 / 64) / 8;  // 8192
  const int gATT = T_TOKENS / 64;                     // 2048

  // ---- swin block 1 (no shift) ----
  k_ln<<<gLN, 256, 0, stream>>>(Y, w_ln_g, w_ln_b, LN, 1, 0);
  k_gemm<0><<<gQKV, 256, 0, stream>>>(LN, WQ0, w_qkv_b, QKV, nullptr,
                                      T_TOKENS, 384, 128, 0);
  k_attn<<<gATT, 256, 0, stream>>>(QKV, ATT, w_bias, 0);
  k_gemm<2><<<gPRJ, 256, 0, stream>>>(ATT, WP0, w_proj_b, nullptr, Y,
                                      T_TOKENS, 128, 128, 0);

  // ---- swin block 2 (shift = 4) ----
  k_ln<<<gLN, 256, 0, stream>>>(Y, s_ln_g, s_ln_b, LN, 1, 4);
  k_gemm<0><<<gQKV, 256, 0, stream>>>(LN, WQ1, s_qkv_b, QKV, nullptr,
                                      T_TOKENS, 384, 128, 0);
  k_attn<<<gATT, 256, 0, stream>>>(QKV, ATT, s_bias, 1);
  k_gemm<2><<<gPRJ, 256, 0, stream>>>(ATT, WP1, s_proj_b, nullptr, Y,
                                      T_TOKENS, 128, 128, 4);

  // ---- FFN ----
  k_ln<<<gLN, 256, 0, stream>>>(Y, f_ln_g, f_ln_b, LN, 0, 0);
  k_gemm<1><<<gFF1, 256, 0, stream>>>(LN, WF1, nullptr, FFH1, nullptr,
                                      T_TOKENS, 512, 128, 0);
  k_dwconv<<<(T_TOKENS * 64) / 256, 256, 0, stream>>>(FFH1, ff_wd, FFH2);
  k_gemm<3><<<2048, 256, 0, stream>>>(FFH2, WF2, nullptr, nullptr, Y,
                                      T_TOKENS, 128, 512, 0);

  // NHWC -> NCHW
  k_nhwc2nchw<<<(T_TOKENS * CC) / 256, 256, 0, stream>>>(Y, (float*)d_out);
}
